// EncoderLayer1_28381143892641
// MI455X (gfx1250) — compile-verified, matrix operand order checked by reading, not run on hardware
//
#include <hip/hip_runtime.h>
#include <hip/hip_bf16.h>

// ---------------------------------------------------------------------------
// EncoderLayer for MI455X (gfx1250, wave32, WMMA).
// All GEMMs (QKV/O projections, FFN) and attention (QK^T, PV) run on
// v_wmma_f32_16x16x32_f16 with f32 accumulation (compute-bound: ~103 GFLOP vs
// ~0.1 GB of traffic, so f16 WMMA is the right precision/path on CDNA5).
//  - Non-transposed LDS tiles are staged with GLOBAL_LOAD_ASYNC_TO_LDS_B128
//    (ASYNCcnt-tracked, bypasses VGPRs) + s_wait_asynccnt.
//  - Double-buffered software pipeline: tile t+1's async copy overlaps
//    tile t's WMMAs; one barrier per k-step.
//  - Softmax cross-lane reductions use VALU DPP butterflies (xor 1/2/7/15 via
//    quad_perm + mirrors), freeing the DS pipe for WMMA fragment loads and
//    co-executing with the matrix pipe.
// Note: MHA without positional terms is permutation-equivariant, so
// flip(MHA(flip(x))) == MHA(x); branch 2 is just MHA with the W2 weights.
// ---------------------------------------------------------------------------

#define BB   8
#define SS   1024
#define NV   768
#define DD   512
#define DFF  2048
#define HH   8
#define DHH  64
#define BSROWS (BB * SS)                       // 8192
#define BSD   ((size_t)BSROWS * DD)            // 4,194,304

typedef _Float16 v16h __attribute__((ext_vector_type(16)));
typedef float    v8f  __attribute__((ext_vector_type(8)));
typedef _Float16 h8   __attribute__((ext_vector_type(8)));

// ---------------- DPP 16-lane butterfly reductions (VALU, no DS pipe) ------
template <int CTRL>
__device__ __forceinline__ float dpp_perm_f(float x) {
  int i = __builtin_bit_cast(int, x);
  int r = __builtin_amdgcn_update_dpp(i, i, CTRL, 0xF, 0xF, true);
  return __builtin_bit_cast(float, r);
}
__device__ __forceinline__ float red16_max(float x) {
  x = fmaxf(x, dpp_perm_f<0xB1>(x));    // quad_perm [1,0,3,2] : xor 1
  x = fmaxf(x, dpp_perm_f<0x4E>(x));    // quad_perm [2,3,0,1] : xor 2
  x = fmaxf(x, dpp_perm_f<0x141>(x));   // row_half_mirror     : xor 7
  x = fmaxf(x, dpp_perm_f<0x140>(x));   // row_mirror          : xor 15
  return x;
}
__device__ __forceinline__ float red16_sum(float x) {
  x += dpp_perm_f<0xB1>(x);
  x += dpp_perm_f<0x4E>(x);
  x += dpp_perm_f<0x141>(x);
  x += dpp_perm_f<0x140>(x);
  return x;
}

// ---------------- async global -> LDS staging (CDNA5, ASYNCcnt) ------------
__device__ __forceinline__ unsigned lds_off_of(const void* p) {
  // generic pointers to LDS carry the offset in the low 32 bits
  return (unsigned)(unsigned long long)(uintptr_t)p;
}
__device__ __forceinline__ void async_load_b128(unsigned lds_off, const void* gaddr) {
  asm volatile("global_load_async_to_lds_b128 %0, %1, off"
               :: "v"(lds_off), "v"((unsigned long long)(uintptr_t)gaddr)
               : "memory");
}
__device__ __forceinline__ void wait_async0() {
  asm volatile("s_wait_asynccnt 0x0" ::: "memory");
}

// ---------------------------------------------------------------------------
// f32 -> f16 convert (weights, etc.)
// ---------------------------------------------------------------------------
__global__ __launch_bounds__(256) void cvt_f16_kernel(const float* __restrict__ src,
                                                      _Float16* __restrict__ dst,
                                                      size_t n) {
  size_t i = (size_t)blockIdx.x * blockDim.x + threadIdx.x;
  size_t stride = (size_t)gridDim.x * blockDim.x;
  for (; i < n; i += stride) dst[i] = (_Float16)src[i];
}

// ---------------------------------------------------------------------------
// Batched gather over first N rows (rest pass-through), optional f16 copy.
// ---------------------------------------------------------------------------
__global__ __launch_bounds__(256) void gather_cast_kernel(const float* __restrict__ x,
                                                          const int* __restrict__ idx,
                                                          float* __restrict__ of,
                                                          _Float16* __restrict__ oh) {
  int bs = blockIdx.x;                 // 0..B*S-1
  int b = bs / SS, s = bs % SS;
  int srow = s;
  if (s < NV) srow = idx[b * NV + s];
  const float* src = x + ((size_t)b * SS + srow) * DD;
  size_t dbase = (size_t)bs * DD;
  for (int d = threadIdx.x; d < DD; d += blockDim.x) {
    float v = src[d];
    of[dbase + d] = v;
    if (oh) oh[dbase + d] = (_Float16)v;
  }
}

// ---------------------------------------------------------------------------
// c = a + b (elementwise f32)
// ---------------------------------------------------------------------------
__global__ __launch_bounds__(256) void add2_kernel(const float* __restrict__ a,
                                                   const float* __restrict__ b,
                                                   float* __restrict__ c, size_t n) {
  size_t i = (size_t)blockIdx.x * blockDim.x + threadIdx.x;
  size_t stride = (size_t)gridDim.x * blockDim.x;
  for (; i < n; i += stride) c[i] = a[i] + b[i];
}

// ---------------------------------------------------------------------------
// out = LayerNorm(a + b) * g + beta, one block (256 thr) per row of D=512.
// ---------------------------------------------------------------------------
__global__ __launch_bounds__(256) void add_ln_kernel(const float* __restrict__ a,
                                                     const float* __restrict__ b,
                                                     const float* __restrict__ g,
                                                     const float* __restrict__ beta,
                                                     float* __restrict__ of,
                                                     _Float16* __restrict__ oh) {
  __shared__ float sred[16];
  int row = blockIdx.x;
  size_t base = (size_t)row * DD;
  int d0 = threadIdx.x, d1 = threadIdx.x + 256;
  float t0 = a[base + d0] + b[base + d0];
  float t1 = a[base + d1] + b[base + d1];
  float s = t0 + t1, ss = t0 * t0 + t1 * t1;
#pragma unroll
  for (int m = 1; m < 32; m <<= 1) { s += __shfl_xor(s, m, 32); ss += __shfl_xor(ss, m, 32); }
  int wave = threadIdx.x >> 5, lane = threadIdx.x & 31;
  if (lane == 0) { sred[wave] = s; sred[8 + wave] = ss; }
  __syncthreads();
  if (wave == 0) {
    float s2 = (lane < 8) ? sred[lane] : 0.f;
    float q2 = (lane < 8) ? sred[8 + lane] : 0.f;
#pragma unroll
    for (int m = 1; m < 8; m <<= 1) { s2 += __shfl_xor(s2, m, 32); q2 += __shfl_xor(q2, m, 32); }
    if (lane == 0) { sred[0] = s2; sred[1] = q2; }
  }
  __syncthreads();
  float mu  = sred[0] * (1.f / DD);
  float var = sred[1] * (1.f / DD) - mu * mu;
  float rstd = rsqrtf(var + 1e-5f);
  float r0 = (t0 - mu) * rstd * g[d0] + beta[d0];
  float r1 = (t1 - mu) * rstd * g[d1] + beta[d1];
  of[base + d0] = r0;
  of[base + d1] = r1;
  if (oh) { oh[base + d0] = (_Float16)r0; oh[base + d1] = (_Float16)r1; }
}

// ---------------------------------------------------------------------------
// WMMA GEMM: C[M,N] = A[M,K] (f16, row-major) @ W[K,N] (f16, row-major)
// Workgroup tile 128x128, 8 waves in a 4(M) x 2(N) grid, each wave 32x64
// (2x4 accumulators of v_wmma_f32_16x16x32_f16). Double-buffered pipeline:
// A-tile via async global->LDS; B-tile transposed through VGPRs, scattered
// into LDS after the current tile's WMMAs. One barrier per k-step.
// ---------------------------------------------------------------------------
template <bool OUT_F16, bool BIAS, bool RELU>
__global__ __launch_bounds__(256) void gemm_wmma(const _Float16* __restrict__ A,
                                                 const _Float16* __restrict__ W,
                                                 const float* __restrict__ bias,
                                                 _Float16* __restrict__ outH,
                                                 float* __restrict__ outF,
                                                 int M, int N, int K) {
  __shared__ _Float16 As[2][128][40];   // ping-pong: 128 rows x 32 K (pad 40)
  __shared__ _Float16 Bt[2][128][40];   // ping-pong: 128 cols(N) x 32 K (transposed)

  const int tid  = threadIdx.x;
  const int lane = tid & 31;
  const int wave = tid >> 5;
  const int l15  = lane & 15;
  const int half = lane >> 4;
  const int wm = wave >> 1;          // 0..3
  const int wn = wave & 1;           // 0..1
  const int mbase = blockIdx.y * 128;
  const int nbase = blockIdx.x * 128;

  v8f acc[2][4];
#pragma unroll
  for (int i = 0; i < 2; i++)
#pragma unroll
    for (int j = 0; j < 4; j++) { v8f z = {}; acc[i][j] = z; }

  const int arow = tid >> 1;            // A tile: row t/2, 16 halfs at (t&1)*16
  const int acs  = (tid & 1) * 16;
  const int bkr  = tid >> 3;            // B tile: k-row t/8, 16 cols at (t&7)*16
  const int bns  = (tid & 7) * 16;

  const int nk = K >> 5;                // k-steps of 32

  {   // prologue: stage tile 0 into buffer 0
    const _Float16* ga = A + (size_t)(mbase + arow) * K + acs;
    async_load_b128(lds_off_of(&As[0][arow][acs]), ga);
    async_load_b128(lds_off_of(&As[0][arow][acs + 8]), ga + 8);
    const h8* src = reinterpret_cast<const h8*>(W + (size_t)bkr * N + nbase + bns);
    h8 w0 = src[0], w1 = src[1];
#pragma unroll
    for (int j = 0; j < 8; j++) Bt[0][bns + j][bkr] = w0[j];
#pragma unroll
    for (int j = 0; j < 8; j++) Bt[0][bns + 8 + j][bkr] = w1[j];
    wait_async0();
    __syncthreads();
  }

  for (int t = 0; t < nk; ++t) {
    const int cur = t & 1, nxt = cur ^ 1;
    const bool more = (t + 1 < nk);
    const int k0n = (t + 1) << 5;
    h8 w0 = {}, w1 = {};
    if (more) {   // issue next tile's copies: async A -> LDS, B -> VGPRs
      const _Float16* ga = A + (size_t)(mbase + arow) * K + k0n + acs;
      async_load_b128(lds_off_of(&As[nxt][arow][acs]), ga);
      async_load_b128(lds_off_of(&As[nxt][arow][acs + 8]), ga + 8);
      const h8* src = reinterpret_cast<const h8*>(W + (size_t)(k0n + bkr) * N + nbase + bns);
      w0 = src[0]; w1 = src[1];
    }

    // ---- compute on current buffers ----
    v16h af[2], bf[4];
#pragma unroll
    for (int mt = 0; mt < 2; mt++) {
      const _Float16* p = &As[cur][wm * 32 + mt * 16 + l15][half * 8];
#pragma unroll
      for (int i = 0; i < 8; i++) { af[mt][i] = p[i]; af[mt][8 + i] = p[16 + i]; }
    }
#pragma unroll
    for (int nt = 0; nt < 4; nt++) {
      const _Float16* p = &Bt[cur][wn * 64 + nt * 16 + l15][half * 8];
#pragma unroll
      for (int i = 0; i < 8; i++) { bf[nt][i] = p[i]; bf[nt][8 + i] = p[16 + i]; }
    }
#pragma unroll
    for (int mt = 0; mt < 2; mt++)
#pragma unroll
      for (int nt = 0; nt < 4; nt++)
        acc[mt][nt] = __builtin_amdgcn_wmma_f32_16x16x32_f16(
            false, af[mt], false, bf[nt], (short)0, acc[mt][nt], false, false);

    if (more) {   // scatter next B tile (overlapped global latency is behind us)
#pragma unroll
      for (int j = 0; j < 8; j++) Bt[nxt][bns + j][bkr] = w0[j];
#pragma unroll
      for (int j = 0; j < 8; j++) Bt[nxt][bns + 8 + j][bkr] = w1[j];
    }
    wait_async0();
    __syncthreads();
  }

  // Epilogue: C layout: VGPR i, lane half h -> row = 8*h + i, col = l15.
#pragma unroll
  for (int mt = 0; mt < 2; mt++) {
#pragma unroll
    for (int nt = 0; nt < 4; nt++) {
      int gc = nbase + wn * 64 + nt * 16 + l15;
      float bv = BIAS ? bias[gc] : 0.f;
#pragma unroll
      for (int i = 0; i < 8; i++) {
        int gr = mbase + wm * 32 + mt * 16 + half * 8 + i;
        float v = acc[mt][nt][i] + bv;
        if (RELU) v = v > 0.f ? v : 0.f;
        size_t off = (size_t)gr * N + gc;
        if (OUT_F16) outH[off] = (_Float16)v;
        else         outF[off] = v;
      }
    }
  }
}

// ---------------------------------------------------------------------------
// Flash-style attention, one workgroup per (128 q-rows, head, batch).
// 8 waves x 16 q-rows; K/V streamed through double-buffered LDS in 32-key
// blocks (K async global->LDS, V transposed through VGPRs). scores: 2 WMMAs
// per 16-key subtile (K=64 contraction); P staged via LDS (C->A layout,
// wave-local s_wait_dscnt); PV: 4 WMMAs. Softmax reductions on VALU DPP.
// ---------------------------------------------------------------------------
__global__ __launch_bounds__(256) void attn_wmma(const _Float16* __restrict__ Q,
                                                 const _Float16* __restrict__ Kk,
                                                 const _Float16* __restrict__ V,
                                                 _Float16* __restrict__ O) {
  __shared__ _Float16 Kb[2][32][72];     // [key][d], pad 64->72
  __shared__ _Float16 Vt[2][64][40];     // [d][key], pad 32->40
  __shared__ _Float16 Pst[8][16][32];    // per-wave P tile [row][key]

  const int tid  = threadIdx.x;
  const int lane = tid & 31;
  const int wave = tid >> 5;
  const int l15  = lane & 15;
  const int half = lane >> 4;
  const int qb   = blockIdx.x * 128;
  const int head = blockIdx.y;
  const int bat  = blockIdx.z;
  const size_t base = (size_t)bat * SS * DD + (size_t)head * DHH;
  const float scale = 0.125f;            // 1/sqrt(64)

  // Q fragments: rows qb + wave*16 + l15, d split {0..31},{32..63}
  v16h qa[2];
  {
    const _Float16* qp = Q + base + (size_t)(qb + wave * 16 + l15) * DD;
#pragma unroll
    for (int f = 0; f < 2; f++)
#pragma unroll
      for (int i = 0; i < 8; i++) {
        qa[f][i]     = qp[f * 32 + half * 8 + i];
        qa[f][8 + i] = qp[f * 32 + 16 + half * 8 + i];
      }
  }

  v8f o[4];
#pragma unroll
  for (int t = 0; t < 4; t++) { v8f z = {}; o[t] = z; }
  float mrun[8], lrun[8];
#pragma unroll
  for (int i = 0; i < 8; i++) { mrun[i] = -1e30f; lrun[i] = 0.f; }

  const int r = tid >> 3;                // 0..31 (key)
  const int c = (tid & 7) * 8;           // 0..56 (d)

  {   // prologue: stage key-block 0 into buffer 0
    async_load_b128(lds_off_of(&Kb[0][r][c]), Kk + base + (size_t)r * DD + c);
    h8 vv = *reinterpret_cast<const h8*>(V + base + (size_t)r * DD + c);
#pragma unroll
    for (int j = 0; j < 8; j++) Vt[0][c + j][r] = vv[j];
    wait_async0();
    __syncthreads();
  }

  const int nkb = SS >> 5;               // 32-key blocks
  for (int t = 0; t < nkb; ++t) {
    const int cur = t & 1, nxt = cur ^ 1;
    const bool more = (t + 1 < nkb);
    h8 vv = {};
    if (more) {   // issue next block's copies
      size_t roff = base + (size_t)((t + 1) * 32 + r) * DD + c;
      async_load_b128(lds_off_of(&Kb[nxt][r][c]), Kk + roff);
      vv = *reinterpret_cast<const h8*>(V + roff);
    }

    // ---- scores: S_tile(16 x 32) in two 16-key subtiles -------------------
    v8f sc[2];
#pragma unroll
    for (int nb = 0; nb < 2; nb++) {
      v8f z = {};
      v16h bf;
      {
        const _Float16* p = &Kb[cur][nb * 16 + l15][half * 8];
#pragma unroll
        for (int i = 0; i < 8; i++) { bf[i] = p[i]; bf[8 + i] = p[16 + i]; }
      }
      z = __builtin_amdgcn_wmma_f32_16x16x32_f16(false, qa[0], false, bf, (short)0, z, false, false);
      {
        const _Float16* p = &Kb[cur][nb * 16 + l15][32 + half * 8];
#pragma unroll
        for (int i = 0; i < 8; i++) { bf[i] = p[i]; bf[8 + i] = p[16 + i]; }
      }
      z = __builtin_amdgcn_wmma_f32_16x16x32_f16(false, qa[1], false, bf, (short)0, z, false, false);
      sc[nb] = z;
    }
#pragma unroll
    for (int nb = 0; nb < 2; nb++)
#pragma unroll
      for (int i = 0; i < 8; i++) sc[nb][i] *= scale;

    // ---- online softmax (row = 8*half + i; DPP reduce across 16 lanes) ----
#pragma unroll
    for (int i = 0; i < 8; i++) {
      float mx = red16_max(fmaxf(sc[0][i], sc[1][i]));
      float mnew = fmaxf(mrun[i], mx);
      float corr = __expf(mrun[i] - mnew);
      float p0 = __expf(sc[0][i] - mnew);
      float p1 = __expf(sc[1][i] - mnew);
      float rs = red16_sum(p0 + p1);
      lrun[i] = lrun[i] * corr + rs;
      mrun[i] = mnew;
#pragma unroll
      for (int tt = 0; tt < 4; tt++) o[tt][i] *= corr;
      Pst[wave][half * 8 + i][l15]      = (_Float16)p0;
      Pst[wave][half * 8 + i][16 + l15] = (_Float16)p1;
    }
    // wave-local LDS RAW: same-wave DS ops are in-order; wait then reload.
    asm volatile("s_wait_dscnt 0x0" ::: "memory");

    // ---- P fragment (A-layout 16x32) --------------------------------------
    v16h pa;
    {
      const _Float16* p = &Pst[wave][l15][half * 8];
#pragma unroll
      for (int i = 0; i < 8; i++) { pa[i] = p[i]; pa[8 + i] = p[16 + i]; }
    }
    // ---- PV: O(16 x 64) += P(16 x 32) @ V(32 x 64) ------------------------
#pragma unroll
    for (int tt = 0; tt < 4; tt++) {
      v16h bf;
      const _Float16* p = &Vt[cur][tt * 16 + l15][half * 8];
#pragma unroll
      for (int i = 0; i < 8; i++) { bf[i] = p[i]; bf[8 + i] = p[16 + i]; }
      o[tt] = __builtin_amdgcn_wmma_f32_16x16x32_f16(false, pa, false, bf, (short)0, o[tt], false, false);
    }

    if (more) {   // scatter next V block into the other buffer
#pragma unroll
      for (int j = 0; j < 8; j++) Vt[nxt][c + j][r] = vv[j];
    }
    wait_async0();
    __syncthreads();
  }

  // ---- normalize and store O ---------------------------------------------
#pragma unroll
  for (int i = 0; i < 8; i++) {
    float inv = 1.f / lrun[i];
    int row = qb + wave * 16 + half * 8 + i;
    _Float16* op = O + base + (size_t)row * DD;
#pragma unroll
    for (int t = 0; t < 4; t++) op[t * 16 + l15] = (_Float16)(o[t][i] * inv);
  }
}

// ---------------------------------------------------------------------------
// Host-side orchestration
// ---------------------------------------------------------------------------
extern "C" void kernel_launch(void* const* d_in, const int* in_sizes, int n_in,
                              void* d_out, int out_size, void* d_ws, size_t ws_size,
                              hipStream_t stream) {
  (void)in_sizes; (void)n_in; (void)out_size; (void)ws_size;

  const float* x        = (const float*)d_in[0];
  const int*   idx_emb  = (const int*)d_in[1];
  const int*   idx_pro  = (const int*)d_in[2];
  // d_in[3] is N == 768 (compile-time constant NV; cannot read device scalar
  // under graph capture).
  const float* Wq1 = (const float*)d_in[4];
  const float* Wk1 = (const float*)d_in[5];
  const float* Wv1 = (const float*)d_in[6];
  const float* Wo1 = (const float*)d_in[7];
  const float* Wq2 = (const float*)d_in[8];
  const float* Wk2 = (const float*)d_in[9];
  const float* Wv2 = (const float*)d_in[10];
  const float* Wo2 = (const float*)d_in[11];
  const float* w_conv1 = (const float*)d_in[12];
  const float* b_conv1 = (const float*)d_in[13];
  const float* w_conv2 = (const float*)d_in[14];
  const float* b_conv2 = (const float*)d_in[15];
  const float* g1 = (const float*)d_in[16];
  const float* beta1 = (const float*)d_in[17];
  const float* g2 = (const float*)d_in[18];
  const float* beta2 = (const float*)d_in[19];

  float* out   = (float*)d_out;          // LN2 result
  float* attn1 = out + BSD;
  float* attn2 = out + 2 * BSD;

  // ---- workspace layout ----
  char* w = (char*)d_ws;
  float* xl1  = (float*)w;  w += BSD * sizeof(float);   // x_l1 f32 (later reused: xln)
  float* f32b = (float*)w;  w += BSD * sizeof(float);   // new_x1 / y2
  float* f32c = (float*)w;  w += BSD * sizeof(float);   // x_l2
  _Float16* h_x = (_Float16*)w; w += BSD * sizeof(_Float16);   // f16 activations
  _Float16* h_q = (_Float16*)w; w += BSD * sizeof(_Float16);
  _Float16* h_k = (_Float16*)w; w += BSD * sizeof(_Float16);
  _Float16* h_v = (_Float16*)w; w += BSD * sizeof(_Float16);
  _Float16* h_o = (_Float16*)w; w += BSD * sizeof(_Float16);
  _Float16* h_ffn = (_Float16*)w; w += (size_t)BSROWS * DFF * sizeof(_Float16);
  _Float16* h_w = (_Float16*)w; w += (size_t)DD * DFF * sizeof(_Float16);   // weight staging

  const dim3 blk(256);
  const dim3 gemmDxD(DD / 128, BSROWS / 128);     // (4, 64)
  const dim3 gemmDxF(DFF / 128, BSROWS / 128);    // (16, 64)
  const dim3 attnGrid(SS / 128, HH, BB);          // (8, 8, 8)

  auto cvt = [&](const float* s, _Float16* d, size_t n) {
    cvt_f16_kernel<<<dim3(2048), blk, 0, stream>>>(s, d, n);
  };

  // 1) x_l1 = gather(x, idx_emb) ; also f16 copy
  gather_cast_kernel<<<dim3(BSROWS), blk, 0, stream>>>(x, idx_emb, xl1, h_x);

  // ================= MHA branch 1 =================
  cvt(Wq1, h_w, (size_t)DD * DD);
  gemm_wmma<true, false, false><<<gemmDxD, blk, 0, stream>>>(h_x, h_w, nullptr, h_q, nullptr, BSROWS, DD, DD);
  cvt(Wk1, h_w, (size_t)DD * DD);
  gemm_wmma<true, false, false><<<gemmDxD, blk, 0, stream>>>(h_x, h_w, nullptr, h_k, nullptr, BSROWS, DD, DD);
  cvt(Wv1, h_w, (size_t)DD * DD);
  gemm_wmma<true, false, false><<<gemmDxD, blk, 0, stream>>>(h_x, h_w, nullptr, h_v, nullptr, BSROWS, DD, DD);
  attn_wmma<<<attnGrid, blk, 0, stream>>>(h_q, h_k, h_v, h_o);
  cvt(Wo1, h_w, (size_t)DD * DD);
  gemm_wmma<false, false, false><<<gemmDxD, blk, 0, stream>>>(h_o, h_w, nullptr, nullptr, attn1, BSROWS, DD, DD);

  // ================= MHA branch 2 (flip∘MHA∘flip == MHA) =================
  cvt(Wq2, h_w, (size_t)DD * DD);
  gemm_wmma<true, false, false><<<gemmDxD, blk, 0, stream>>>(h_x, h_w, nullptr, h_q, nullptr, BSROWS, DD, DD);
  cvt(Wk2, h_w, (size_t)DD * DD);
  gemm_wmma<true, false, false><<<gemmDxD, blk, 0, stream>>>(h_x, h_w, nullptr, h_k, nullptr, BSROWS, DD, DD);
  cvt(Wv2, h_w, (size_t)DD * DD);
  gemm_wmma<true, false, false><<<gemmDxD, blk, 0, stream>>>(h_x, h_w, nullptr, h_v, nullptr, BSROWS, DD, DD);
  attn_wmma<<<attnGrid, blk, 0, stream>>>(h_q, h_k, h_v, h_o);
  cvt(Wo2, h_w, (size_t)DD * DD);
  gemm_wmma<false, false, false><<<gemmDxD, blk, 0, stream>>>(h_o, h_w, nullptr, nullptr, attn2, BSROWS, DD, DD);

  // new_x1 = attn1 + attn2 ; x_l2 = gather(new_x1, idx_pro)
  add2_kernel<<<dim3(4096), blk, 0, stream>>>(attn1, attn2, f32b, BSD);
  gather_cast_kernel<<<dim3(BSROWS), blk, 0, stream>>>(f32b, idx_pro, f32c, nullptr);

  // xln = LN(x + x_l2; g1, beta1)  (reuse xl1 as f32 result, h_x as f16 copy)
  add_ln_kernel<<<dim3(BSROWS), blk, 0, stream>>>(x, f32c, g1, beta1, xl1, h_x);

  // ================= FFN =================
  cvt(w_conv1, h_w, (size_t)DD * DFF);
  gemm_wmma<true, true, true><<<gemmDxF, blk, 0, stream>>>(h_x, h_w, b_conv1, h_ffn, nullptr, BSROWS, DFF, DD);
  cvt(w_conv2, h_w, (size_t)DFF * DD);
  gemm_wmma<false, true, false><<<gemmDxD, blk, 0, stream>>>(h_ffn, h_w, b_conv2, nullptr, f32b, BSROWS, DD, DFF);

  // out = LN(xln + y; g2, beta2)
  add_ln_kernel<<<dim3(BSROWS), blk, 0, stream>>>(xl1, f32b, g2, beta2, out, nullptr);
}
